// EncoderBlock_61400852463678
// MI455X (gfx1250) — matrix operand order repaired and checked by reading, not compiled, not run on hardware
//
#include <hip/hip_runtime.h>
#include <hip/hip_bf16.h>

// ---- problem dims (fixed by reference) ----
#define TB 2
#define TS 2048
#define TE 1024
#define TH 8
#define TF 4
#define NEGINF  (-1000000.0f)
#define LN_EPS  1e-5f

typedef __attribute__((ext_vector_type(16))) _Float16 v16h;
typedef __attribute__((ext_vector_type(8)))  _Float16 v8h;
typedef __attribute__((ext_vector_type(8)))  float    v8f;

// ---------------- CDNA5 async global->LDS helpers ----------------
// GLOBAL_LOAD_ASYNC_TO_LDS_B128 (GV mode): vdst = LDS byte offset VGPR,
// vaddr = 64-bit global address VGPR pair. Tracked by ASYNCcnt.
__device__ __forceinline__ void async_ld_b128(unsigned lds_off, unsigned long long gaddr) {
  asm volatile("global_load_async_to_lds_b128 %0, %1, off"
               :: "v"(lds_off), "v"(gaddr) : "memory");
}
__device__ __forceinline__ void wait_async0() {
  asm volatile("s_wait_asynccnt 0" ::: "memory");
}
// Low 32 bits of a generic LDS pointer == wave-relative LDS byte offset (ISA 10.2 aperture rules).
__device__ __forceinline__ unsigned lds_off_of(const void* p) {
  return (unsigned)(unsigned long long)(uintptr_t)p;
}

// ---------------- elementwise f32 -> f16 ----------------
__global__ void k_cvt_f16(const float* __restrict__ in, _Float16* __restrict__ out, long long n) {
  long long i = (long long)blockIdx.x * blockDim.x + threadIdx.x;
  long long stride = (long long)gridDim.x * blockDim.x;
  for (; i < n; i += stride) out[i] = (_Float16)in[i];
}

// ------- batched transpose+convert: f32 (R x C) -> f16 (C x R) -------
__global__ void k_transpose_cvt(const float* __restrict__ in, _Float16* __restrict__ out,
                                int R, int C) {
  __shared__ float t[32][33];
  long long zoff = (long long)blockIdx.z * R * C;
  in += zoff; out += zoff;
  int c = blockIdx.x * 32 + threadIdx.x;
  #pragma unroll
  for (int k = 0; k < 4; ++k) {
    int r = blockIdx.y * 32 + threadIdx.y + k * 8;
    t[threadIdx.y + k * 8][threadIdx.x] = in[(long long)r * C + c];
  }
  __syncthreads();
  int r2 = blockIdx.y * 32 + threadIdx.x;            // output column = original row
  #pragma unroll
  for (int k = 0; k < 4; ++k) {
    int c2 = blockIdx.x * 32 + threadIdx.y + k * 8;  // output row = original col
    out[(long long)c2 * R + r2] = (_Float16)t[threadIdx.x][threadIdx.y + k * 8];
  }
}

// ---------------- WMMA GEMM with async->LDS double-buffered staging ----------------
// C[m][n] = alpha * sum_k A[m][k] * Bt[n][k]  (+bias[n]) (+relu)
// 128 threads = 4 waves; block tile 128x128; each wave computes a 64x64 quadrant
// as a 4x4 grid of v_wmma_f32_16x16x32_f16. K staged in LDS, 32 per step,
// double-buffered via GLOBAL_LOAD_ASYNC_TO_LDS_B128 + s_wait_asynccnt.
__global__ __launch_bounds__(128) void k_gemm(
    const _Float16* __restrict__ A, const _Float16* __restrict__ Bt,
    _Float16* __restrict__ C16, float* __restrict__ C32,
    const float* __restrict__ bias,
    int M, int N, int K, int lda, int ldb, int ldc,
    int Hdim,
    long long sAb, long long sAh, long long sBb, long long sBh,
    long long sCb, long long sCh,
    float alpha, int relu)
{
  __shared__ __align__(16) _Float16 Alds[2][128][32];   // 8KB per buffer
  __shared__ __align__(16) _Float16 Blds[2][128][32];   // 8KB per buffer

  int z = blockIdx.z;
  int bb = z / Hdim, hh = z % Hdim;
  A  += bb * sAb + hh * sAh;
  Bt += bb * sBb + hh * sBh;
  long long coff = bb * sCb + hh * sCh;

  int m0 = blockIdx.y * 128;
  int n0 = blockIdx.x * 128;
  int tid  = threadIdx.x;
  int lane = tid & 31;
  int wave = tid >> 5;
  int l15  = lane & 15;
  int hi   = lane >> 4;              // 0: lanes 0-15, 1: lanes 16-31
  int wm   = (wave >> 1) * 64;       // wave quadrant inside the 128x128 tile
  int wn   = (wave & 1) * 64;

  // cooperative staging: thread t owns row t of both tiles (4 x b128 each)
  unsigned long long gA = (unsigned long long)(uintptr_t)(A  + (long long)(m0 + tid) * lda);
  unsigned long long gB = (unsigned long long)(uintptr_t)(Bt + (long long)(n0 + tid) * ldb);
  unsigned aBase = lds_off_of(&Alds[0][tid][0]);
  unsigned bBase = lds_off_of(&Blds[0][tid][0]);
  const unsigned bufStride = 128 * 32 * 2;             // bytes per buffer

  auto issue = [&](int buf, int k0) {
    unsigned ao = aBase + (unsigned)buf * bufStride;
    unsigned bo = bBase + (unsigned)buf * bufStride;
    unsigned long long ga = gA + 2ull * (unsigned)k0;
    unsigned long long gb = gB + 2ull * (unsigned)k0;
    #pragma unroll
    for (int c = 0; c < 4; ++c) {
      async_ld_b128(ao + c * 16, ga + c * 16ull);
      async_ld_b128(bo + c * 16, gb + c * 16ull);
    }
  };

  v8f acc[4][4] = {};
  int nsteps = K / 32;
  issue(0, 0);

  for (int s = 0; s < nsteps; ++s) {
    int buf = s & 1;
    wait_async0();          // this wave's share of buf is in LDS
    __syncthreads();        // everyone's share of buf is in LDS
    if (s + 1 < nsteps) issue((s + 1) & 1, (s + 1) * 32);  // overlap next DMA with compute

    v16h af[4], bf[4];
    #pragma unroll
    for (int i = 0; i < 4; ++i) {
      // A frag: lanes 0-15 hold K {0..7,16..23}; lanes 16-31 hold K {8..15,24..31}
      const _Float16* p = &Alds[buf][wm + 16 * i + l15][hi * 8];
      v8h lo = *(const v8h*)(p);
      v8h hv = *(const v8h*)(p + 16);
      af[i] = __builtin_shufflevector(lo, hv, 0,1,2,3,4,5,6,7,8,9,10,11,12,13,14,15);
    }
    #pragma unroll
    for (int j = 0; j < 4; ++j) {
      // B frag: lanes 0-15 hold K {0..7,16..23}; lanes 16-31 hold K {8..15,24..31}
      const _Float16* p = &Blds[buf][wn + 16 * j + l15][hi * 8];
      v8h lo = *(const v8h*)(p);
      v8h hv = *(const v8h*)(p + 16);
      bf[j] = __builtin_shufflevector(lo, hv, 0,1,2,3,4,5,6,7,8,9,10,11,12,13,14,15);
    }
    #pragma unroll
    for (int i = 0; i < 4; ++i)
      #pragma unroll
      for (int j = 0; j < 4; ++j)
        acc[i][j] = __builtin_amdgcn_wmma_f32_16x16x32_f16(
            false, af[i], false, bf[j], (short)0, acc[i][j], false, false);

    __syncthreads();        // all waves done reading buf before it is overwritten
  }

  // epilogue: C/D layout -> VGPR r holds rows (r, r+8), col = lane&15
  #pragma unroll
  for (int i = 0; i < 4; ++i) {
    #pragma unroll
    for (int j = 0; j < 4; ++j) {
      #pragma unroll
      for (int r = 0; r < 8; ++r) {
        int row = m0 + wm + 16 * i + r + hi * 8;
        int col = n0 + wn + 16 * j + l15;
        float v = acc[i][j][r] * alpha;
        if (bias) v += bias[col];
        if (relu) v = v > 0.f ? v : 0.f;
        long long off = coff + (long long)row * ldc + col;
        if (C32) C32[off] = v;
        if (C16) C16[off] = (_Float16)v;
      }
    }
  }
}

// ---------------- masked softmax over rows of length S ----------------
__global__ __launch_bounds__(256) void k_softmax(
    float* __restrict__ scores, _Float16* __restrict__ w16,
    const int* __restrict__ mask)
{
  __shared__ float red[256];
  int row = blockIdx.x;                 // 0 .. B*H*S-1
  int tid = threadIdx.x;
  int b = row / (TH * TS);
  int q = row % TS;
  float* sc = scores + (long long)row * TS;
  const int* mrow = mask + ((long long)b * TS + q) * TS;

  float vals[TS / 256];
  float mx = -3.4e38f;
  #pragma unroll
  for (int t = 0; t < TS / 256; ++t) {
    int idx = tid + t * 256;
    float v = sc[idx];
    if (mrow[idx] == 0) v = NEGINF;
    vals[t] = v;
    mx = fmaxf(mx, v);
  }
  red[tid] = mx; __syncthreads();
  for (int s = 128; s > 0; s >>= 1) { if (tid < s) red[tid] = fmaxf(red[tid], red[tid + s]); __syncthreads(); }
  mx = red[0]; __syncthreads();

  float sum = 0.f;
  #pragma unroll
  for (int t = 0; t < TS / 256; ++t) { vals[t] = expf(vals[t] - mx); sum += vals[t]; }
  red[tid] = sum; __syncthreads();
  for (int s = 128; s > 0; s >>= 1) { if (tid < s) red[tid] += red[tid + s]; __syncthreads(); }
  float inv = 1.f / red[0];

  #pragma unroll
  for (int t = 0; t < TS / 256; ++t) {
    int idx = tid + t * 256;
    float w = vals[t] * inv;
    sc[idx] = w;
    w16[(long long)row * TS + idx] = (_Float16)w;
  }
}

// ---------------- residual add + layernorm over rows of length E ----------------
__global__ __launch_bounds__(256) void k_add_ln(
    const float* __restrict__ x, const float* __restrict__ r,
    const float* __restrict__ w, const float* __restrict__ b,
    float* __restrict__ out32, _Float16* __restrict__ out16)
{
  __shared__ float red[256];
  long long base = (long long)blockIdx.x * TE;
  int tid = threadIdx.x;
  float v[TE / 256];
  float s = 0.f;
  #pragma unroll
  for (int t = 0; t < TE / 256; ++t) {
    int idx = tid + t * 256;
    v[t] = x[base + idx] + r[base + idx];
    s += v[t];
  }
  red[tid] = s; __syncthreads();
  for (int st = 128; st > 0; st >>= 1) { if (tid < st) red[tid] += red[tid + st]; __syncthreads(); }
  float mean = red[0] / TE; __syncthreads();

  float s2 = 0.f;
  #pragma unroll
  for (int t = 0; t < TE / 256; ++t) { float d = v[t] - mean; s2 += d * d; }
  red[tid] = s2; __syncthreads();
  for (int st = 128; st > 0; st >>= 1) { if (tid < st) red[tid] += red[tid + st]; __syncthreads(); }
  float inv = rsqrtf(red[0] / TE + LN_EPS);

  #pragma unroll
  for (int t = 0; t < TE / 256; ++t) {
    int idx = tid + t * 256;
    float g = (v[t] - mean) * inv * w[idx] + b[idx];
    out32[base + idx] = g;
    if (out16) out16[base + idx] = (_Float16)g;
  }
}

extern "C" void kernel_launch(void* const* d_in, const int* in_sizes, int n_in,
                              void* d_out, int out_size, void* d_ws, size_t ws_size,
                              hipStream_t stream) {
  (void)in_sizes; (void)n_in; (void)out_size; (void)ws_size;

  const float* X    = (const float*)d_in[0];
  const int*   mask = (const int*)d_in[1];
  const float* Wq   = (const float*)d_in[2];
  const float* Wk   = (const float*)d_in[3];
  const float* Wv   = (const float*)d_in[4];
  const float* Wz   = (const float*)d_in[5];
  const float* bz   = (const float*)d_in[6];
  const float* W1   = (const float*)d_in[7];
  const float* b1   = (const float*)d_in[8];
  const float* W2   = (const float*)d_in[9];
  const float* b2   = (const float*)d_in[10];
  const float* ln1w = (const float*)d_in[11];
  const float* ln1b = (const float*)d_in[12];
  const float* ln2w = (const float*)d_in[13];
  const float* ln2b = (const float*)d_in[14];

  const long long NX   = (long long)TB * TS * TE;           // 4.19M
  const long long NHSE = (long long)TB * TH * TS * TE;      // 33.6M
  const long long NZ   = (long long)TB * TS * TH * TE;      // 33.6M
  const long long NHID = (long long)TB * TS * TF * TE;      // 16.8M

  float* outZ    = (float*)d_out;              // (B,S,E)
  float* outAttn = outZ + NX;                  // (B,H,S,S) — also f32 score scratch

  // ---- workspace layout (256B aligned bump allocator) ----
  char* ws = (char*)d_ws;
  size_t off = 0;
  auto alloc = [&](size_t bytes) -> char* {
    char* p = ws + off;
    off += (bytes + 255) & ~(size_t)255;
    return p;
  };
  _Float16* Xh   = (_Float16*)alloc(NX * 2);
  _Float16* WqT  = (_Float16*)alloc((size_t)TH * TE * TE * 2);
  _Float16* WkT  = (_Float16*)alloc((size_t)TH * TE * TE * 2);
  _Float16* WvT  = (_Float16*)alloc((size_t)TH * TE * TE * 2);
  _Float16* WzT  = (_Float16*)alloc((size_t)TH * TE * TE * 2);
  _Float16* W1T  = (_Float16*)alloc((size_t)TE * TF * TE * 2);
  _Float16* W2T  = (_Float16*)alloc((size_t)TE * TF * TE * 2);
  _Float16* Q    = (_Float16*)alloc(NHSE * 2);
  _Float16* Kf   = (_Float16*)alloc(NHSE * 2);   // contiguous after Q
  _Float16* Vt   = (_Float16*)alloc(NHSE * 2);
  _Float16* AW16 = Q;   // alias: B*H*S*S f16 == |Q|+|K| exactly (S == 2E); Q,K dead after scores
  _Float16* Zcat = (_Float16*)alloc(NZ * 2);
  float*    y0   = (float*)alloc(NX * 4);
  float*    y1f  = (float*)alloc(NX * 4);
  _Float16* y1h  = (_Float16*)alloc(NX * 2);
  _Float16* hid  = (_Float16*)alloc(NHID * 2);
  float*    ff   = (float*)alloc(NX * 4);

  const float scale = 0.17677669529663687f;  // 1024^(-1/4)
  dim3 tbG(128), tb256(256), tbT(32, 8);

  // 1) X -> f16
  k_cvt_f16<<<dim3(4096), tb256, 0, stream>>>(X, Xh, NX);

  // 2) transpose+convert all weights
  k_transpose_cvt<<<dim3(TE/32, TE/32, TH), tbT, 0, stream>>>(Wq, WqT, TE, TE);
  k_transpose_cvt<<<dim3(TE/32, TE/32, TH), tbT, 0, stream>>>(Wk, WkT, TE, TE);
  k_transpose_cvt<<<dim3(TE/32, TE/32, TH), tbT, 0, stream>>>(Wv, WvT, TE, TE);
  k_transpose_cvt<<<dim3(TE/32, (TH*TE)/32, 1), tbT, 0, stream>>>(Wz, WzT, TH*TE, TE);
  k_transpose_cvt<<<dim3((TF*TE)/32, TE/32, 1), tbT, 0, stream>>>(W1, W1T, TE, TF*TE);
  k_transpose_cvt<<<dim3(TE/32, (TF*TE)/32, 1), tbT, 0, stream>>>(W2, W2T, TF*TE, TE);

  // 3) Q = scale * X @ Wq[h]   (B,H,S,E) f16
  k_gemm<<<dim3(TE/128, TS/128, TB*TH), tbG, 0, stream>>>(
      Xh, WqT, Q, nullptr, nullptr,
      TS, TE, TE, TE, TE, TE, TH,
      (long long)TS*TE, 0LL, 0LL, (long long)TE*TE,
      (long long)TH*TS*TE, (long long)TS*TE, scale, 0);
  // 4) K = scale * X @ Wk[h]
  k_gemm<<<dim3(TE/128, TS/128, TB*TH), tbG, 0, stream>>>(
      Xh, WkT, Kf, nullptr, nullptr,
      TS, TE, TE, TE, TE, TE, TH,
      (long long)TS*TE, 0LL, 0LL, (long long)TE*TE,
      (long long)TH*TS*TE, (long long)TS*TE, scale, 0);
  // 5) V^T = Wv[h]^T @ X^T   -> (B,H,E,S) f16
  k_gemm<<<dim3(TS/128, TE/128, TB*TH), tbG, 0, stream>>>(
      WvT, Xh, Vt, nullptr, nullptr,
      TE, TS, TE, TE, TE, TS, TH,
      0LL, (long long)TE*TE, (long long)TS*TE, 0LL,
      (long long)TH*TE*TS, (long long)TE*TS, 1.f, 0);
  // 6) scores = Q @ K^T  -> f32 straight into d_out attn region
  k_gemm<<<dim3(TS/128, TS/128, TB*TH), tbG, 0, stream>>>(
      Q, Kf, nullptr, outAttn, nullptr,
      TS, TS, TE, TE, TE, TS, TH,
      (long long)TH*TS*TE, (long long)TS*TE, (long long)TH*TS*TE, (long long)TS*TE,
      (long long)TH*TS*TS, (long long)TS*TS, 1.f, 0);
  // 7) masked softmax in-place (f32 output) + f16 copy (overlays dead Q||K)
  k_softmax<<<dim3(TB*TH*TS), tb256, 0, stream>>>(outAttn, AW16, mask);
  // 8) Zcat[b,s,h*E+e] = attn_w @ V
  k_gemm<<<dim3(TE/128, TS/128, TB*TH), tbG, 0, stream>>>(
      AW16, Vt, Zcat, nullptr, nullptr,
      TS, TE, TS, TS, TS, TH*TE, TH,
      (long long)TH*TS*TS, (long long)TS*TS, (long long)TH*TE*TS, (long long)TE*TS,
      (long long)TS*TH*TE, (long long)TE, 1.f, 0);
  // 9) y0 = Zcat @ Wz + bz   (f32)
  k_gemm<<<dim3(TE/128, (TB*TS)/128, 1), tbG, 0, stream>>>(
      Zcat, WzT, nullptr, y0, bz,
      TB*TS, TE, TH*TE, TH*TE, TH*TE, TE, 1,
      0LL, 0LL, 0LL, 0LL, 0LL, 0LL, 1.f, 0);
  // 10) y1 = LN1(y0 + X)  (f32 + f16)
  k_add_ln<<<dim3(TB*TS), tb256, 0, stream>>>(y0, X, ln1w, ln1b, y1f, y1h);
  // 11) hid = relu(y1 @ W1 + b1)  (f16)
  k_gemm<<<dim3((TF*TE)/128, (TB*TS)/128, 1), tbG, 0, stream>>>(
      y1h, W1T, hid, nullptr, b1,
      TB*TS, TF*TE, TE, TE, TE, TF*TE, 1,
      0LL, 0LL, 0LL, 0LL, 0LL, 0LL, 1.f, 1);
  // 12) ff = hid @ W2 + b2  (f32)
  k_gemm<<<dim3(TE/128, (TB*TS)/128, 1), tbG, 0, stream>>>(
      hid, W2T, nullptr, ff, b2,
      TB*TS, TE, TF*TE, TF*TE, TF*TE, TE, 1,
      0LL, 0LL, 0LL, 0LL, 0LL, 0LL, 1.f, 0);
  // 13) z = LN2(ff + y1) -> d_out
  k_add_ln<<<dim3(TB*TS), tb256, 0, stream>>>(ff, y1f, ln2w, ln2b, outZ, nullptr);
}
